// GATPressureGNN_27762668601578
// MI455X (gfx1250) — matrix operand order repaired and checked by reading, not compiled, not run on hardware
//
#include <hip/hip_runtime.h>
#include <hip/hip_bf16.h>
#include <cstddef>

#define NNODES 100000
#define NEDGES 1600000
#define ETOT   (NEDGES + NNODES)
#define NEG_SLOPE 0.2f
#define EPW 4   // edges per wave in aggregation

typedef __attribute__((ext_vector_type(16))) __bf16 v16bf;
typedef __attribute__((ext_vector_type(8)))  float  v8f;

// f32 -> bf16 bits, round-to-nearest-even
static __device__ __forceinline__ unsigned bfbits(float f) {
    union { float f; unsigned u; } cv; cv.f = f;
    unsigned u = cv.u;
    u += 0x7FFFu + ((u >> 16) & 1u);
    return u >> 16;
}

static __device__ __forceinline__ __bf16 f2bf(float f) {
    unsigned short hs = (unsigned short)bfbits(f);
    __bf16 r;
    __builtin_memcpy(&r, &hs, sizeof(r));
    return r;
}

// C[M x 64] = A[M x K] * B[K x 64]  (+bias, +relu), bf16 WMMA, wave-per-16-row tile.
//
// B is converted f32->bf16 and SWIZZLED into WMMA-fragment order once per block:
// for fragment (kt,nt), lane L, element e (B-frag layout: col N = L&15,
// K = kt*32 + e + (L<16 ? 0 : 16)), the bf16 value lives at packed-dword index
//   frag*256 + (e>>3)*128 + L*4 + ((e>>1)&3)   (low/high half by e&1)
// so each fragment build is exactly two ds_load_b128 per lane.
//
// A-frag layout used here (16-bit A 16x32): lane<16 -> K = 0..15, lane>=16 -> K = 16..31 (same order as B)
// C/D f32 16x16: VGPR r -> M = r + (L<16 ? 0 : 8), N = L&15
template<int K, bool RELU, bool BIAS>
__global__ __launch_bounds__(256)
void gat_wmma_gemm(const float* __restrict__ A,
                   const float* __restrict__ B,
                   const float* __restrict__ bias,
                   float* __restrict__ C, int M) {
    __shared__ unsigned Bsp[K * 32];             // K*64 bf16 elements, fragment-swizzled
    {
        for (int i = threadIdx.x; i < K * 32; i += 256) {
            int n   = i & 63;                    // column (coalesced across threads)
            int k0  = (i >> 6) * 2;              // even row; pair (k0, k0+1)
            int kt  = k0 >> 5;
            int kk  = k0 & 31;
            int col = n & 15;
            int ls  = (kk < 16) ? col : col + 16;    // target lane
            int e   = kk & 15;                        // even element index
            int frag = kt * 4 + (n >> 4);
            int idx = frag * 256 + ((e >> 3) * 128) + ls * 4 + ((e >> 1) & 3);
            unsigned lo = bfbits(B[(size_t)k0 * 64 + n]);
            unsigned hi = bfbits(B[(size_t)(k0 + 1) * 64 + n]);
            Bsp[idx] = lo | (hi << 16);
        }
    }
    __syncthreads();

    const int lane  = threadIdx.x & 31;
    const int wave  = threadIdx.x >> 5;
    const int mtile = blockIdx.x * 8 + wave;
    if (mtile * 16 >= M) return;

    const bool hi      = lane >= 16;
    const int  col     = lane & 15;
    const int  kbase_a = hi ? 16 : 0;
    const int  row     = mtile * 16 + col;

    v8f acc[4] = {};

#pragma unroll
    for (int kt = 0; kt < K / 32; ++kt) {
        const float4* a4 = (const float4*)(A + (size_t)row * K + kt * 32 + kbase_a);
        float4 f0 = a4[0], f1 = a4[1], f2 = a4[2], f3 = a4[3];
        float tmp[16] = {f0.x,f0.y,f0.z,f0.w, f1.x,f1.y,f1.z,f1.w,
                         f2.x,f2.y,f2.z,f2.w, f3.x,f3.y,f3.z,f3.w};
        v16bf a;
#pragma unroll
        for (int e = 0; e < 16; ++e) a[e] = f2bf(tmp[e]);

#pragma unroll
        for (int nt = 0; nt < 4; ++nt) {
            const uint4* fb = (const uint4*)(Bsp + (kt * 4 + nt) * 256 + lane * 4);
            uint4 blo = fb[0];                   // elements e = 0..7
            uint4 bhi = fb[32];                  // elements e = 8..15 (+128 dwords)
            union { unsigned u[8]; v16bf v; } bb;
            bb.u[0] = blo.x; bb.u[1] = blo.y; bb.u[2] = blo.z; bb.u[3] = blo.w;
            bb.u[4] = bhi.x; bb.u[5] = bhi.y; bb.u[6] = bhi.z; bb.u[7] = bhi.w;
            acc[nt] = __builtin_amdgcn_wmma_f32_16x16x32_bf16(
                false, a, false, bb.v, (short)0, acc[nt], false, false);
        }
    }

    const int r0 = hi ? 8 : 0;
#pragma unroll
    for (int nt = 0; nt < 4; ++nt) {
        const int n = nt * 16 + col;
#pragma unroll
        for (int r = 0; r < 8; ++r) {
            float v = acc[nt][r];
            if (BIAS) v += bias[n];
            if (RELU) v = fmaxf(v, 0.0f);
            C[(size_t)(mtile * 16 + r0 + r) * 64 + n] = v;
        }
    }
}

// Per (node, head): alpha_src/alpha_dst dot products; zero z and agg.
__global__ __launch_bounds__(256)
void k_alpha_init(const float* __restrict__ hw, const float* __restrict__ att_s,
                  const float* __restrict__ att_d, float* __restrict__ as_,
                  float* __restrict__ ad_, float* __restrict__ z,
                  float* __restrict__ agg) {
    int t = blockIdx.x * 256 + threadIdx.x;
    if (t >= NNODES * 4) return;
    int n = t >> 2, hd = t & 3;
    const float4* hp = (const float4*)(hw + (size_t)n * 64 + hd * 16);
    const float4* sp = (const float4*)(att_s + hd * 16);
    const float4* dp = (const float4*)(att_d + hd * 16);
    float s = 0.f, d = 0.f;
#pragma unroll
    for (int i = 0; i < 4; ++i) {
        float4 hv = hp[i], sv = sp[i], dv = dp[i];
        s += hv.x * sv.x + hv.y * sv.y + hv.z * sv.z + hv.w * sv.w;
        d += hv.x * dv.x + hv.y * dv.y + hv.z * dv.z + hv.w * dv.w;
    }
    as_[t] = s; ad_[t] = d; z[t] = 0.f;
    float4* ag = (float4*)(agg + (size_t)n * 64 + hd * 16);
    float4 z4 = make_float4(0.f, 0.f, 0.f, 0.f);
#pragma unroll
    for (int i = 0; i < 4; ++i) ag[i] = z4;
}

// Per (edge, head): softmax denominator (segment-max elided: e is bounded, exp safe,
// softmax is shift-invariant so result matches reference up to fp rounding).
__global__ __launch_bounds__(256)
void k_edge_denom(const int* __restrict__ eidx, const float* __restrict__ as_,
                  const float* __restrict__ ad_, float* __restrict__ z) {
    int t = blockIdx.x * 256 + threadIdx.x;
    if (t >= ETOT * 4) return;
    int ei = t >> 2, hd = t & 3;
    int src, dst;
    if (ei < NEDGES) { src = eidx[ei]; dst = eidx[NEDGES + ei]; }
    else             { src = dst = ei - NEDGES; }
    float e = as_[src * 4 + hd] + ad_[dst * 4 + hd];
    e = (e > 0.f) ? e : NEG_SLOPE * e;
    atomicAdd(&z[dst * 4 + hd], __expf(e));
}

// Aggregation: 4 edges per wave, 2 channels per lane. Indices loaded up front and
// gathered rows prefetched (global_prefetch_b8) to hide random-access latency,
// then coalesced float2 gather + 2 f32 atomics per lane per edge.
__global__ __launch_bounds__(256)
void k_edge_agg(const int* __restrict__ eidx, const float* __restrict__ as_,
                const float* __restrict__ ad_, const float* __restrict__ z,
                const float* __restrict__ hw, float* __restrict__ agg) {
    int t = blockIdx.x * 256 + threadIdx.x;
    int wid = t >> 5;
    int ebase = wid * EPW;
    if (ebase >= ETOT) return;      // ETOT % EPW == 0: all EPW edges below are valid
    int lane = t & 31;
    int hd = lane >> 3;             // 8 lanes (16 channels) per head
    int c  = lane * 2;

    int srcs[EPW], dsts[EPW];
#pragma unroll
    for (int i = 0; i < EPW; ++i) {
        int ei = ebase + i;
        if (ei < NEDGES) { srcs[i] = eidx[ei]; dsts[i] = eidx[NEDGES + ei]; }
        else             { srcs[i] = dsts[i] = ei - NEDGES; }
    }
#pragma unroll
    for (int i = 0; i < EPW; ++i)
        __builtin_prefetch(hw + (size_t)srcs[i] * 64 + c, 0, 0);

#pragma unroll
    for (int i = 0; i < EPW; ++i) {
        int src = srcs[i], dst = dsts[i];
        float e = as_[src * 4 + hd] + ad_[dst * 4 + hd];
        e = (e > 0.f) ? e : NEG_SLOPE * e;
        float alpha = __expf(e) / (z[dst * 4 + hd] + 1e-16f);
        float2 hv = *(const float2*)(hw + (size_t)src * 64 + c);
        atomicAdd(&agg[(size_t)dst * 64 + c],     alpha * hv.x);
        atomicAdd(&agg[(size_t)dst * 64 + c + 1], alpha * hv.y);
    }
}

// Wave-per-node: bias + LayerNorm (shuffle reduction) + ReLU + residual, in-place h.
__global__ __launch_bounds__(256)
void k_post(const float* __restrict__ agg, const float* __restrict__ bg,
            const float* __restrict__ g, const float* __restrict__ b,
            float* __restrict__ h) {
    int t = blockIdx.x * 256 + threadIdx.x;
    int node = t >> 5;
    if (node >= NNODES) return;
    int lane = t & 31;
    int c = lane * 2;
    float2 v = *(const float2*)(agg + (size_t)node * 64 + c);
    v.x += bg[c]; v.y += bg[c + 1];
    float s  = v.x + v.y;
    float sq = v.x * v.x + v.y * v.y;
#pragma unroll
    for (int m = 16; m >= 1; m >>= 1) {
        s  += __shfl_xor(s,  m, 32);
        sq += __shfl_xor(sq, m, 32);
    }
    float mu  = s  * (1.0f / 64.0f);
    float var = sq * (1.0f / 64.0f) - mu * mu;
    float inv = rsqrtf(var + 1e-5f);
    float2 ho = *(const float2*)(h + (size_t)node * 64 + c);
    float2 r;
    r.x = fmaxf((v.x - mu) * inv * g[c]     + b[c],     0.f) + ho.x;
    r.y = fmaxf((v.y - mu) * inv * g[c + 1] + b[c + 1], 0.f) + ho.y;
    *(float2*)(h + (size_t)node * 64 + c) = r;
}

// Final projection: out[n] = h[n] . Wout + bout
__global__ __launch_bounds__(256)
void k_out(const float* __restrict__ h, const float* __restrict__ Wout,
           const float* __restrict__ bout, float* __restrict__ out) {
    int n = blockIdx.x * 256 + threadIdx.x;
    if (n >= NNODES) return;
    const float4* hp = (const float4*)(h + (size_t)n * 64);
    const float4* wp = (const float4*)Wout;
    float acc = bout[0];
#pragma unroll
    for (int i = 0; i < 16; ++i) {
        float4 hv = hp[i], wv = wp[i];
        acc += hv.x * wv.x + hv.y * wv.y + hv.z * wv.z + hv.w * wv.w;
    }
    out[n] = acc;
}

extern "C" void kernel_launch(void* const* d_in, const int* in_sizes, int n_in,
                              void* d_out, int out_size, void* d_ws, size_t ws_size,
                              hipStream_t stream) {
    const float* x       = (const float*)d_in[0];
    const int*   eidx    = (const int*)  d_in[1];
    const float* Win     = (const float*)d_in[2];
    const float* bin_    = (const float*)d_in[3];
    const float* Wg      = (const float*)d_in[4];
    const float* att_src = (const float*)d_in[5];
    const float* att_dst = (const float*)d_in[6];
    const float* bg      = (const float*)d_in[7];
    const float* ln_g    = (const float*)d_in[8];
    const float* ln_b    = (const float*)d_in[9];
    const float* Wout    = (const float*)d_in[10];
    const float* bout    = (const float*)d_in[11];
    float* out = (float*)d_out;

    float* ws  = (float*)d_ws;
    float* h   = ws;                                 // N*64
    float* hw  = ws + (size_t)NNODES * 64;           // N*64
    float* agg = ws + (size_t)NNODES * 128;          // N*64
    float* as_ = ws + (size_t)NNODES * 192;          // N*4
    float* ad_ = as_ + (size_t)NNODES * 4;           // N*4
    float* zs  = ad_ + (size_t)NNODES * 4;           // N*4

    dim3 blk(256);
    const int gemm_grid = (NNODES + 127) / 128;      // 8 wave-tiles of 16 rows per block
    const int agg_grid  = (ETOT / EPW + 7) / 8;      // 8 waves per block, EPW edges per wave

    // h = relu(x @ Win + bin)
    gat_wmma_gemm<128, true, true><<<gemm_grid, blk, 0, stream>>>(x, Win, bin_, h, NNODES);

    for (int l = 0; l < 3; ++l) {
        // hw = h @ Wg[l]
        gat_wmma_gemm<64, false, false><<<gemm_grid, blk, 0, stream>>>(
            h, Wg + (size_t)l * 64 * 64, nullptr, hw, NNODES);
        // alpha_src/dst per (node,head); zero z, agg
        k_alpha_init<<<(NNODES * 4 + 255) / 256, blk, 0, stream>>>(
            hw, att_src + l * 64, att_dst + l * 64, as_, ad_, zs, agg);
        // softmax denominators
        k_edge_denom<<<(ETOT * 4 + 255) / 256, blk, 0, stream>>>(eidx, as_, ad_, zs);
        // weighted message aggregation (4 edges per wave, prefetch-pipelined)
        k_edge_agg<<<agg_grid, blk, 0, stream>>>(eidx, as_, ad_, zs, hw, agg);
        // bias + LN + relu + residual -> h (in place)
        k_post<<<(NNODES + 7) / 8, blk, 0, stream>>>(
            agg, bg + l * 64, ln_g + l * 64, ln_b + l * 64, h);
    }

    k_out<<<(NNODES + 255) / 256, blk, 0, stream>>>(h, Wout, bout, out);
}
